// Qwen3MoeSparseMoeBlock_67851893342706
// MI455X (gfx1250) — matrix operand order repaired and checked by reading, not compiled, not run on hardware
//
#include <hip/hip_runtime.h>

#define T_TOK 1024
#define H_DIM 2048
#define E_NUM 32
#define I_DIM 768
#define TOPK  8
#define CAP   512
#define BK    32
#define RS    40   // LDS row stride in bf16 elems (32 + 8 pad -> 80B rows, conflict-free 16B frag loads)

typedef __attribute__((ext_vector_type(16))) __bf16 v16bf;
typedef __attribute__((ext_vector_type(8)))  __bf16 v8bf;
typedef __attribute__((ext_vector_type(8)))  float  v8f;
typedef __attribute__((ext_vector_type(4)))  float  v4f;

// ---------------- helpers ----------------

// Split 16 fp32 into bf16 hi/lo planes and store 16B-vectorized into LDS.
__device__ __forceinline__ void cvt_store16(__bf16* __restrict__ hiP, __bf16* __restrict__ loP,
                                            v4f x0, v4f x1, v4f x2, v4f x3) {
  float xs[16];
#pragma unroll
  for (int i = 0; i < 4; i++) { xs[i] = x0[i]; xs[4 + i] = x1[i]; xs[8 + i] = x2[i]; xs[12 + i] = x3[i]; }
  v8bf h0, h1, l0, l1;
#pragma unroll
  for (int i = 0; i < 8; i++) {
    float a = xs[i];
    __bf16 ah = (__bf16)a;           // RNE
    h0[i] = ah;
    l0[i] = (__bf16)(a - (float)ah); // residual, ~2^-9 of a
    float b = xs[8 + i];
    __bf16 bh = (__bf16)b;
    h1[i] = bh;
    l1[i] = (__bf16)(b - (float)bh);
  }
  *(v8bf*)(hiP)     = h0;
  *(v8bf*)(hiP + 8) = h1;
  *(v8bf*)(loP)     = l0;
  *(v8bf*)(loP + 8) = l1;
}

// A fragment (16x32 bf16): lane = M (lane&15); lanes<16 hold K{0..7,16..23}, lanes>=16 hold K{8..15,24..31}
__device__ __forceinline__ v16bf frag_load_A(const __bf16* base, int row, int lane) {
  const __bf16* p = base + row * RS + ((lane >> 4) << 3);
  v8bf a = *(const v8bf*)p;          // K k0..k0+7
  v8bf b = *(const v8bf*)(p + 16);   // K k0+16..k0+23
  v16bf r;
#pragma unroll
  for (int i = 0; i < 8; i++) { r[i] = a[i]; r[8 + i] = b[i]; }
  return r;
}

// B fragment (32x16 bf16): lane = N; lanes<16 hold K{0..7,16..23}, lanes>=16 hold K{8..15,24..31}
__device__ __forceinline__ v16bf frag_load_B(const __bf16* base, int row, int lane) {
  const __bf16* p = base + row * RS + ((lane >> 4) << 3);
  v8bf a = *(const v8bf*)p;
  v8bf b = *(const v8bf*)(p + 16);
  v16bf r;
#pragma unroll
  for (int i = 0; i < 8; i++) { r[i] = a[i]; r[8 + i] = b[i]; }
  return r;
}

// fp32-accurate product via bf16 split: hi*hi + hi*lo + lo*hi
__device__ __forceinline__ v8f wmma3(v16bf ah, v16bf al, v16bf bh, v16bf bl, v8f c) {
  c = __builtin_amdgcn_wmma_f32_16x16x32_bf16(false, ah, false, bh, (short)0, c, false, false);
  c = __builtin_amdgcn_wmma_f32_16x16x32_bf16(false, ah, false, bl, (short)0, c, false, false);
  c = __builtin_amdgcn_wmma_f32_16x16x32_bf16(false, al, false, bh, (short)0, c, false, false);
  return c;
}

// ---------------- kernel 0: init ----------------
__global__ __launch_bounds__(256) void moe_init_kernel(float* __restrict__ out, int* __restrict__ counts) {
  int i = blockIdx.x * blockDim.x + threadIdx.x;
  if (i < T_TOK * H_DIM) out[i] = 0.f;
  if (i < E_NUM) counts[i] = 0;
}

// ---------------- kernel 1: router (softmax + top-8 + slot assignment) ----------------
__global__ __launch_bounds__(32) void moe_router_kernel(
    const float* __restrict__ hidden, const float* __restrict__ w_gate,
    int* __restrict__ counts, int* __restrict__ slot_token, float* __restrict__ slot_w) {
  const int t = blockIdx.x;
  const int lane = threadIdx.x;
  const float* hrow = hidden + (size_t)t * H_DIM;

  float logit[E_NUM];
#pragma unroll
  for (int e = 0; e < E_NUM; e++) {
    const float* wrow = w_gate + (size_t)e * H_DIM;
    float s = 0.f;
#pragma unroll 4
    for (int j = 0; j < H_DIM / 32; j++) {
      int h = lane + 32 * j;
      s = fmaf(hrow[h], wrow[h], s);
    }
#pragma unroll
    for (int off = 16; off >= 1; off >>= 1) s += __shfl_xor(s, off, 32);
    logit[e] = s;  // identical in every lane
  }

  // fp32 softmax (replicated per lane)
  float mx = logit[0];
#pragma unroll
  for (int e = 1; e < E_NUM; e++) mx = fmaxf(mx, logit[e]);
  float p[E_NUM];
  float den = 0.f;
#pragma unroll
  for (int e = 0; e < E_NUM; e++) { p[e] = expf(logit[e] - mx); den += p[e]; }
  float inv_den = 1.f / den;
#pragma unroll
  for (int e = 0; e < E_NUM; e++) p[e] *= inv_den;

  // top-8 via chosen-mask (strict '>' keeps lowest index on ties, matching lax.top_k)
  unsigned chosen = 0u;
  int   sel8[TOPK];
  float w8[TOPK];
  float wsum = 0.f;
#pragma unroll
  for (int k = 0; k < TOPK; k++) {
    float best = -1.f;
    int   bi = 0;
#pragma unroll
    for (int e = 0; e < E_NUM; e++) {
      bool c = (((chosen >> e) & 1u) == 0u) && (p[e] > best);
      best = c ? p[e] : best;
      bi   = c ? e : bi;
    }
    chosen |= (1u << bi);
    sel8[k] = bi;
    w8[k]   = best;
    wsum   += best;
  }

  if (lane == 0) {
    float inv = 1.f / wsum;
#pragma unroll
    for (int k = 0; k < TOPK; k++) {
      int e = sel8[k];
      int pos = atomicAdd(&counts[e], 1);
      if (pos < CAP) {
        slot_token[e * CAP + pos] = t;
        slot_w[e * CAP + pos]     = w8[k] * inv;
      }
    }
  }
}

// ---------------- kernel 2: GEMM1 + SwiGLU  (xbuf @ gate_upT -> hmid) ----------------
// grid: E * 4(mtiles) * 12(ntiles of 64 hmid cols); block 256 (8 waves)
__global__ __launch_bounds__(256) void moe_gemm1_kernel(
    const float* __restrict__ hidden, const float* __restrict__ gate_up,
    const int* __restrict__ counts, const int* __restrict__ slot_token,
    float* __restrict__ hmid) {
  __shared__ __bf16 sAh[128 * RS], sAl[128 * RS], sBh[128 * RS], sBl[128 * RS];

  const int bx = blockIdx.x;
  const int e   = bx / 48;
  const int r48 = bx % 48;
  const int mt  = r48 / 12;
  const int nt  = r48 % 12;
  int cnt = counts[e];
  if (cnt > CAP) cnt = CAP;
  if (mt * 128 >= cnt) return;

  const int tid  = threadIdx.x;
  const int lane = tid & 31;
  const int wave = tid >> 5;
  const int lrow = tid >> 1;          // loader row 0..127
  const int lcol = (tid & 1) * 16;    // 0 or 16

  // A: gathered token rows
  const int mglob = mt * 128 + lrow;
  const float* aptr = nullptr;
  if (mglob < cnt) {
    int tok = slot_token[e * CAP + mglob];
    aptr = hidden + (size_t)tok * H_DIM + lcol;
  }
  // B: 64 gate rows + 64 up rows of gate_up_proj[e]
  const int n0 = nt * 64;
  const int grow = (lrow < 64) ? (n0 + lrow) : (I_DIM + n0 + (lrow - 64));
  const float* bptr = gate_up + ((size_t)e * (2 * I_DIM) + grow) * H_DIM + lcol;

  v8f acc[8];
#pragma unroll
  for (int i = 0; i < 8; i++) acc[i] = v8f{0.f, 0.f, 0.f, 0.f, 0.f, 0.f, 0.f, 0.f};

  const int mrow = wave * 16 + (lane & 15);  // frag row for this wave's 16-row strip

  for (int kb = 0; kb < H_DIM; kb += BK) {
    v4f a0 = v4f{0, 0, 0, 0}, a1 = a0, a2 = a0, a3 = a0;
    if (aptr) {
      const v4f* p = (const v4f*)(aptr + kb);
      a0 = p[0]; a1 = p[1]; a2 = p[2]; a3 = p[3];
    }
    const v4f* q = (const v4f*)(bptr + kb);
    v4f b0 = q[0], b1 = q[1], b2 = q[2], b3 = q[3];

    if (kb + BK < H_DIM) {
      __builtin_prefetch(bptr + kb + BK, 0, 0);
      if (aptr) __builtin_prefetch(aptr + kb + BK, 0, 0);
    }

    __syncthreads();  // protect previous iteration's frag reads
    cvt_store16(&sAh[lrow * RS + lcol], &sAl[lrow * RS + lcol], a0, a1, a2, a3);
    cvt_store16(&sBh[lrow * RS + lcol], &sBl[lrow * RS + lcol], b0, b1, b2, b3);
    __syncthreads();

    v16bf ah = frag_load_A(sAh, mrow, lane);
    v16bf al = frag_load_A(sAl, mrow, lane);
#pragma unroll
    for (int t = 0; t < 8; t++) {
      int brow = t * 16 + (lane & 15);
      v16bf bh = frag_load_B(sBh, brow, lane);
      v16bf bl = frag_load_B(sBl, brow, lane);
      acc[t] = wmma3(ah, al, bh, bl, acc[t]);
    }
  }

  // SwiGLU epilogue: tiles 0..3 = gate cols, 4..7 = matching up cols
  const int m_in = wave * 16 + ((lane >> 4) << 3);
  const int colb = n0 + (lane & 15);
#pragma unroll
  for (int t = 0; t < 4; t++) {
#pragma unroll
    for (int r = 0; r < 8; r++) {
      float g = acc[t][r];
      float u = acc[t + 4][r];
      float h = (g / (1.f + expf(-g))) * u;  // silu(g)*u
      int m = mt * 128 + m_in + r;
      hmid[(size_t)(e * CAP + m) * I_DIM + colb + t * 16] = h;
    }
  }
}

// ---------------- kernel 3: GEMM2 + weighted scatter-add  (hmid @ downT -> out) ----------------
// grid: E * 4(mtiles) * 16(ntiles of 128 H cols); block 256 (8 waves)
__global__ __launch_bounds__(256) void moe_gemm2_kernel(
    const float* __restrict__ hmid, const float* __restrict__ down,
    const int* __restrict__ counts, const int* __restrict__ slot_token,
    const float* __restrict__ slot_w, float* __restrict__ out) {
  __shared__ __bf16 sAh[128 * RS], sAl[128 * RS], sBh[128 * RS], sBl[128 * RS];

  const int bx = blockIdx.x;
  const int e   = bx / 64;
  const int r64 = bx % 64;
  const int mt  = r64 / 16;
  const int nt  = r64 % 16;
  int cnt = counts[e];
  if (cnt > CAP) cnt = CAP;
  if (mt * 128 >= cnt) return;

  const int tid  = threadIdx.x;
  const int lane = tid & 31;
  const int wave = tid >> 5;
  const int lrow = tid >> 1;
  const int lcol = (tid & 1) * 16;

  const float* aptr = hmid + (size_t)(e * CAP + mt * 128 + lrow) * I_DIM + lcol;
  const int n0 = nt * 128;
  const float* bptr = down + ((size_t)e * H_DIM + (n0 + lrow)) * I_DIM + lcol;

  v8f acc[8];
#pragma unroll
  for (int i = 0; i < 8; i++) acc[i] = v8f{0.f, 0.f, 0.f, 0.f, 0.f, 0.f, 0.f, 0.f};

  const int mrow = wave * 16 + (lane & 15);

  for (int kb = 0; kb < I_DIM; kb += BK) {
    const v4f* pa = (const v4f*)(aptr + kb);
    v4f a0 = pa[0], a1 = pa[1], a2 = pa[2], a3 = pa[3];
    const v4f* pb = (const v4f*)(bptr + kb);
    v4f b0 = pb[0], b1 = pb[1], b2 = pb[2], b3 = pb[3];

    if (kb + BK < I_DIM) {
      __builtin_prefetch(aptr + kb + BK, 0, 0);
      __builtin_prefetch(bptr + kb + BK, 0, 0);
    }

    __syncthreads();
    cvt_store16(&sAh[lrow * RS + lcol], &sAl[lrow * RS + lcol], a0, a1, a2, a3);
    cvt_store16(&sBh[lrow * RS + lcol], &sBl[lrow * RS + lcol], b0, b1, b2, b3);
    __syncthreads();

    v16bf ah = frag_load_A(sAh, mrow, lane);
    v16bf al = frag_load_A(sAl, mrow, lane);
#pragma unroll
    for (int t = 0; t < 8; t++) {
      int brow = t * 16 + (lane & 15);
      v16bf bh = frag_load_B(sBh, brow, lane);
      v16bf bl = frag_load_B(sBl, brow, lane);
      acc[t] = wmma3(ah, al, bh, bl, acc[t]);
    }
  }

  // combine: out[token] += w * obuf_row  (scatter-add; rows >= cnt are skipped)
  const int m_in = wave * 16 + ((lane >> 4) << 3);
#pragma unroll
  for (int t = 0; t < 8; t++) {
    int col = n0 + t * 16 + (lane & 15);
#pragma unroll
    for (int r = 0; r < 8; r++) {
      int m = mt * 128 + m_in + r;
      if (m < cnt) {
        int   tok = slot_token[e * CAP + m];
        float w   = slot_w[e * CAP + m];
        atomicAdd(&out[(size_t)tok * H_DIM + col], w * acc[t][r]);
      }
    }
  }
}

// ---------------- launch ----------------
extern "C" void kernel_launch(void* const* d_in, const int* in_sizes, int n_in,
                              void* d_out, int out_size, void* d_ws, size_t ws_size,
                              hipStream_t stream) {
  (void)in_sizes; (void)n_in; (void)out_size; (void)ws_size;
  const float* hidden  = (const float*)d_in[0];
  const float* w_gate  = (const float*)d_in[1];
  const float* gate_up = (const float*)d_in[2];
  const float* down    = (const float*)d_in[3];
  float* out = (float*)d_out;

  char* ws = (char*)d_ws;
  int*   counts     = (int*)(ws + 0);                            // 32 ints (256B reserved)
  int*   slot_token = (int*)(ws + 256);                          // E*CAP ints
  float* slot_w     = (float*)(ws + 256 + E_NUM * CAP * 4);      // E*CAP floats
  float* hmid       = (float*)(ws + 256 + 2 * E_NUM * CAP * 4);  // E*CAP*I floats (~50 MB)

  moe_init_kernel<<<(T_TOK * H_DIM + 255) / 256, 256, 0, stream>>>(out, counts);
  moe_router_kernel<<<T_TOK, 32, 0, stream>>>(hidden, w_gate, counts, slot_token, slot_w);
  moe_gemm1_kernel<<<E_NUM * 4 * 12, 256, 0, stream>>>(hidden, gate_up, counts, slot_token, hmid);
  moe_gemm2_kernel<<<E_NUM * 4 * 16, 256, 0, stream>>>(hmid, down, counts, slot_token, slot_w, out);
}
